// ManifoldConstrainedHyperConnection_549755814396
// MI455X (gfx1250) — matrix operand order repaired and checked by reading, not compiled, not run on hardware
//
#include <hip/hip_runtime.h>
#include <hip/hip_bf16.h>
#include <hip/hip_fp16.h>

typedef __attribute__((ext_vector_type(16))) _Float16 v16h;
typedef __attribute__((ext_vector_type(8)))  float    v8f;

#define N_STREAMS 4
#define S_DIM     4096
#define B_DIM     2
#define D_DIM     2048
#define ND        8192                 // n*D features
#define SBD       (S_DIM*B_DIM*D_DIM)  // 16777216 elems per stream
#define NPOS      (S_DIM*B_DIM)        // 8192 positions
#define NCOL      32                   // 28 alpha + 4 beta (padded GEMM N)
#define KCHUNKS   (ND/32)              // 256 k-chunks of 32

// itertools.permutations(range(4)) lexicographic order
__device__ const unsigned char PERM4[24][4] = {
  {0,1,2,3},{0,1,3,2},{0,2,1,3},{0,2,3,1},{0,3,1,2},{0,3,2,1},
  {1,0,2,3},{1,0,3,2},{1,2,0,3},{1,2,3,0},{1,3,0,2},{1,3,2,0},
  {2,0,1,3},{2,0,3,1},{2,1,0,3},{2,1,3,0},{2,3,0,1},{2,3,1,0},
  {3,0,1,2},{3,0,2,1},{3,1,0,2},{3,1,2,0},{3,2,0,1},{3,2,1,0}
};

__device__ __forceinline__ float4 f4fma(float s, float4 a, float4 acc) {
  acc.x = fmaf(s, a.x, acc.x); acc.y = fmaf(s, a.y, acc.y);
  acc.z = fmaf(s, a.z, acc.z); acc.w = fmaf(s, a.w, acc.w);
  return acc;
}

// ---------------------------------------------------------------------------
// Pack Weff = (1 + norm_weight[f]) * W[f, col] into f16 WMMA-B lane layout.
// bpack layout: [(chunk*2 + ntile)*32 + lane] -> v16h fragment (32B / lane).
// ISA B layout (32x16, 16-bit, wave32): lane L holds column N = L%16; its 16
// packed halves hold K = (e&7) + 8*(L>=16) + 16*(e>=8) (V0..V3 <-> K 0..7, V4..V7 <-> K 16..23
// for lanes 0-15; K 8..15, 24..31 for lanes 16-31).
// col = ntile*16 + (L&15);  col<28 -> dynamic_alpha_fn, else dynamic_beta_fn.
// ---------------------------------------------------------------------------
__global__ __launch_bounds__(256)
void mchc_pack_b(const float* __restrict__ nw,
                 const float* __restrict__ afn,   // [ND, 28]
                 const float* __restrict__ bfn,   // [ND, 4]
                 _Float16* __restrict__ bpack)
{
  int idx = blockIdx.x * 256 + threadIdx.x;       // 0 .. 262143
  int e    = idx & 15;
  int lane = (idx >> 4) & 31;
  int t    = (idx >> 9) & 1;
  int kc   = idx >> 10;                           // 0..255
  int K    = (e & 7) + ((lane >> 4) << 3) + ((e >> 3) << 4); // table-order B layout (matches A)
  int f    = kc * 32 + K;
  int col  = t * 16 + (lane & 15);
  float w  = (col < 28) ? afn[f * 28 + col] : bfn[f * 4 + (col - 28)];
  bpack[idx] = (_Float16)((1.0f + nw[f]) * w);
}

// ---------------------------------------------------------------------------
// Main kernel: one block of 256 threads (8 waves) per 16-position tile.
// ---------------------------------------------------------------------------
__global__ __launch_bounds__(256)
void mchc_main(const float* __restrict__ X,      // [4, S, B, D]
               const float* __restrict__ xout,   // [S, B, D]
               const float* __restrict__ sa,     // [28]
               const float* __restrict__ sb,     // [4]
               const float* __restrict__ ps, const float* __restrict__ rs,
               const float* __restrict__ hs,
               const _Float16* __restrict__ bpack,
               float* __restrict__ out)          // [5, S, B, D]
{
  __shared__ float lds_c[2 * 256];   // two 16x16 f32 C tiles (cross-wave acc)
  __shared__ float lds_ss[16];       // per-row sum of squares
  __shared__ float lds_g[16 * 24];   // per-row gates: H[16], alpha[4], beta[4]

  const int tid  = threadIdx.x;
  const int wave = tid >> 5;
  const int lane = tid & 31;
  const int row  = lane & 15;
  const int tile = blockIdx.x;               // 0..511
  const int p    = tile * 16 + row;          // this lane's A-row position

  for (int i = tid; i < 512 + 16; i += 256) {
    if (i < 512) lds_c[i] = 0.0f; else lds_ss[i - 512] = 0.0f;
  }
  __syncthreads();

  // ================= Phase 1: [16 x 8192] @ [8192 x 32] via WMMA ==========
  v8f c0 = {}; v8f c1 = {};
  float ssq = 0.0f;
  const int    kwave = wave * (ND / 8);      // 1024 features per wave
  const size_t pD    = (size_t)p * D_DIM;
  const int    half8 = (lane >> 4) << 3;     // 0 or 8 (A-fragment K offset)
  const v16h*  bp    = (const v16h*)bpack;

  #pragma unroll 2
  for (int c = 0; c < 32; ++c) {
    // ---- A fragment: 16 raw f32 features of this lane's row -> f16 ----
    const int f0 = kwave + c * 32 + half8;   // 8-feature run (same stream j)
    const int f1 = f0 + 16;                  // second 8-feature run
    const float* r0 = X + (size_t)(f0 >> 11) * SBD + pD + (f0 & 2047);
    const float* r1 = X + (size_t)(f1 >> 11) * SBD + pD + (f1 & 2047);
    float4 a0 = *(const float4*)(r0);
    float4 a1 = *(const float4*)(r0 + 4);
    float4 a2 = *(const float4*)(r1);
    float4 a3 = *(const float4*)(r1 + 4);

    ssq = fmaf(a0.x,a0.x, fmaf(a0.y,a0.y, fmaf(a0.z,a0.z, fmaf(a0.w,a0.w, ssq))));
    ssq = fmaf(a1.x,a1.x, fmaf(a1.y,a1.y, fmaf(a1.z,a1.z, fmaf(a1.w,a1.w, ssq))));
    ssq = fmaf(a2.x,a2.x, fmaf(a2.y,a2.y, fmaf(a2.z,a2.z, fmaf(a2.w,a2.w, ssq))));
    ssq = fmaf(a3.x,a3.x, fmaf(a3.y,a3.y, fmaf(a3.z,a3.z, fmaf(a3.w,a3.w, ssq))));

    v16h a;
    a[0] =(_Float16)a0.x; a[1] =(_Float16)a0.y; a[2] =(_Float16)a0.z; a[3] =(_Float16)a0.w;
    a[4] =(_Float16)a1.x; a[5] =(_Float16)a1.y; a[6] =(_Float16)a1.z; a[7] =(_Float16)a1.w;
    a[8] =(_Float16)a2.x; a[9] =(_Float16)a2.y; a[10]=(_Float16)a2.z; a[11]=(_Float16)a2.w;
    a[12]=(_Float16)a3.x; a[13]=(_Float16)a3.y; a[14]=(_Float16)a3.z; a[15]=(_Float16)a3.w;

    // ---- B fragments (pre-packed, L2-resident) ----
    const int cg = (wave << 5) + c;          // global k-chunk id
    v16h b0v = bp[(cg * 2 + 0) * 32 + lane];
    v16h b1v = bp[(cg * 2 + 1) * 32 + lane];
    if (c + 1 < 32) {                        // prefetch next chunk's B
      __builtin_prefetch((const void*)&bp[(cg * 2 + 2) * 32 + lane], 0, 3);
    }

    c0 = __builtin_amdgcn_wmma_f32_16x16x32_f16(false, a, false, b0v,
                                                (short)0, c0, false, false);
    c1 = __builtin_amdgcn_wmma_f32_16x16x32_f16(false, a, false, b1v,
                                                (short)0, c1, false, false);
  }

  // sum-of-squares: combine the two lane halves (same row), then cross-wave
  ssq += __shfl_xor(ssq, 16);
  if (lane < 16) atomicAdd(&lds_ss[row], ssq);
  #pragma unroll
  for (int v = 0; v < 8; ++v) {
    atomicAdd(&lds_c[v * 32 + lane], c0[v]);
    atomicAdd(&lds_c[256 + v * 32 + lane], c1[v]);
  }
  __syncthreads();

  // ================= Gates: one thread per row ============================
  if (tid < 16) {
    const int r = tid;
    const float rsq = rsqrtf(lds_ss[r] * (1.0f / (float)ND) + 1e-6f);
    const float pre = ps[0], res = rs[0], hp = hs[0];

    float wc[32];
    #pragma unroll
    for (int cc = 0; cc < 32; ++cc) {
      int t = cc >> 4, nl = cc & 15;
      int ln = nl + ((r >> 3) << 4);
      wc[cc] = lds_c[t * 256 + (r & 7) * 32 + ln] * rsq;  // C[M=r, N=nl]
    }
    // alpha_pre = sigmoid(pre*wc[:4] + static_alpha[:4])
    float al[4];
    #pragma unroll
    for (int i = 0; i < 4; ++i)
      al[i] = 1.0f / (1.0f + __expf(-(fmaf(pre, wc[i], sa[i]))));
    // res_coeff = softmax(res*wc[4:28] + static_alpha[4:28])
    float lg[24], m = -3.0e38f;
    #pragma unroll
    for (int k = 0; k < 24; ++k) { lg[k] = fmaf(res, wc[4 + k], sa[4 + k]); m = fmaxf(m, lg[k]); }
    float sum = 0.0f;
    #pragma unroll
    for (int k = 0; k < 24; ++k) { lg[k] = __expf(lg[k] - m); sum += lg[k]; }
    const float inv = 1.0f / sum;
    float H[16];
    #pragma unroll
    for (int i = 0; i < 16; ++i) H[i] = 0.0f;
    #pragma unroll
    for (int k = 0; k < 24; ++k) {
      const float cf = lg[k] * inv;
      #pragma unroll
      for (int i = 0; i < 4; ++i) H[i * 4 + PERM4[k][i]] += cf;
    }
    // beta = 2*sigmoid(hp*wc[28:32] + static_beta)
    float be[4];
    #pragma unroll
    for (int i = 0; i < 4; ++i)
      be[i] = 2.0f / (1.0f + __expf(-(fmaf(hp, wc[28 + i], sb[i]))));

    #pragma unroll
    for (int i = 0; i < 16; ++i) lds_g[r * 24 + i] = H[i];
    #pragma unroll
    for (int i = 0; i < 4; ++i) { lds_g[r * 24 + 16 + i] = al[i]; lds_g[r * 24 + 20 + i] = be[i]; }
  }
  __syncthreads();

  // ================= Phase 2: streaming gated outputs (exact f32) =========
  for (int r = 0; r < 16; ++r) {
    float H[16], al[4], be[4];
    #pragma unroll
    for (int i = 0; i < 16; ++i) H[i] = lds_g[r * 24 + i];
    #pragma unroll
    for (int i = 0; i < 4; ++i) { al[i] = lds_g[r * 24 + 16 + i]; be[i] = lds_g[r * 24 + 20 + i]; }

    const size_t qD = (size_t)(tile * 16 + r) * D_DIM;
    const float4* x0 = (const float4*)(X + qD);
    const float4* x1 = (const float4*)(X + (size_t)SBD + qD);
    const float4* x2 = (const float4*)(X + 2 * (size_t)SBD + qD);
    const float4* x3 = (const float4*)(X + 3 * (size_t)SBD + qD);
    const float4* xo = (const float4*)(xout + qD);

    #pragma unroll
    for (int it = 0; it < 2; ++it) {
      const int i4 = tid + it * 256;              // 512 float4 per row
      float4 v0 = x0[i4], v1 = x1[i4], v2 = x2[i4], v3 = x3[i4], vo = xo[i4];

      float4 z = {0.f, 0.f, 0.f, 0.f};
      float4 br = f4fma(al[3], v3, f4fma(al[2], v2, f4fma(al[1], v1, f4fma(al[0], v0, z))));
      ((float4*)(out + qD))[i4] = br;

      #pragma unroll
      for (int i = 0; i < 4; ++i) {
        float4 h = f4fma(be[i], vo,
                   f4fma(H[i*4+3], v3,
                   f4fma(H[i*4+2], v2,
                   f4fma(H[i*4+1], v1,
                   f4fma(H[i*4+0], v0, z)))));
        ((float4*)(out + (size_t)(1 + i) * SBD + qD))[i4] = h;
      }
    }
  }
}

extern "C" void kernel_launch(void* const* d_in, const int* in_sizes, int n_in,
                              void* d_out, int out_size, void* d_ws, size_t ws_size,
                              hipStream_t stream) {
  const float* X    = (const float*)d_in[0];
  const float* xout = (const float*)d_in[1];
  const float* nw   = (const float*)d_in[2];
  const float* afn  = (const float*)d_in[3];
  const float* sa   = (const float*)d_in[4];
  const float* bfn  = (const float*)d_in[5];
  const float* sb   = (const float*)d_in[6];
  const float* ps   = (const float*)d_in[7];
  const float* rs   = (const float*)d_in[8];
  const float* hs   = (const float*)d_in[9];
  float* out        = (float*)d_out;
  _Float16* bpack   = (_Float16*)d_ws;     // 256 chunks * 2 tiles * 512 f16 = 512 KB

  // Pack folded RMSNorm-weight * projection into WMMA-B f16 layout (L2-hot).
  mchc_pack_b<<<(KCHUNKS * 2 * 32 * 16) / 256, 256, 0, stream>>>(nw, afn, bfn, bpack);
  // 512 tiles of 16 positions each.
  mchc_main<<<NPOS / 16, 256, 0, stream>>>(X, xout, sa, sb, ps, rs, hs, bpack, out);
}
